// GraphAttentionLayer_19344532702056
// MI455X (gfx1250) — matrix operand order repaired and checked by reading, not compiled, not run on hardware
//
#include <hip/hip_runtime.h>

#define NN    8192
#define INF   512
#define OUTF  256
#define LRA   0.2f

typedef __attribute__((ext_vector_type(2)))  float  v2f;
typedef __attribute__((ext_vector_type(8)))  float  v8f;
typedef __attribute__((ext_vector_type(16))) __bf16 v16bf;

static __device__ __forceinline__ unsigned short f2bf(float f) {
  unsigned int u = __float_as_uint(f);
  u += 0x7FFFu + ((u >> 16) & 1u);            // round-to-nearest-even
  return (unsigned short)(u >> 16);
}

// ---------------------------------------------------------------------------
// Kernel 1: Wh = h @ W in full fp32 via V_WMMA_F32_16X16X4_F32.
// Also writes WhbT = bf16(Wh)^T  [OUTF][NN]  so the attention GEMM's
// B-operand has its K dimension contiguous per lane.
// Block = 256 thr = 8 waves; block owns a 16-row tile, wave w owns cols
// [32w, 32w+32) as two 16x16 WMMA tiles.
// ---------------------------------------------------------------------------
__global__ void __launch_bounds__(256)
k_gemm_wh(const float* __restrict__ h, const float* __restrict__ W,
          float* __restrict__ Wh, unsigned short* __restrict__ WhbT)
{
  const int i0   = blockIdx.x * 16;
  const int wave = threadIdx.x >> 5;
  const int lane = threadIdx.x & 31;
  const int n0   = wave * 32;
  const int lm   = lane & 15;
  const int hi   = lane >> 4;                  // 0: lanes 0-15, 1: lanes 16-31

  v8f c0 = {}; v8f c1 = {};
  const int arow = i0 + lm;
  for (int k = 0; k < INF; k += 4) {
    // A 16x4 f32 frag: lanes0-15 hold K=k,k+1 ; lanes16-31 hold K=k+2,k+3
    v2f a = *(const v2f*)(h + arow * INF + k + hi * 2);
    const int kr = k + hi * 2;
    v2f b0, b1;                                // B 4x16 frags, tiles n0 / n0+16
    b0.x = W[kr * OUTF + n0 + lm];       b0.y = W[(kr + 1) * OUTF + n0 + lm];
    b1.x = W[kr * OUTF + n0 + 16 + lm];  b1.y = W[(kr + 1) * OUTF + n0 + 16 + lm];
    c0 = __builtin_amdgcn_wmma_f32_16x16x4_f32(false, a, false, b0, (short)0, c0, false, false);
    c1 = __builtin_amdgcn_wmma_f32_16x16x4_f32(false, a, false, b1, (short)0, c1, false, false);
  }
  #pragma unroll
  for (int r = 0; r < 8; ++r) {
    const int m  = i0 + r + hi * 8;            // C layout: lanes0-15 M=r, lanes16-31 M=8+r
    const int nA = n0 + lm, nB = n0 + 16 + lm;
    Wh[m * OUTF + nA] = c0[r];
    Wh[m * OUTF + nB] = c1[r];
    WhbT[(long)nA * NN + m] = f2bf(c0[r]);
    WhbT[(long)nB * NN + m] = f2bf(c1[r]);
  }
}

// ---------------------------------------------------------------------------
// Kernel 2: s1 = Wh @ a[:OUTF], s2 = Wh @ a[OUTF:].  One wave per row.
// ---------------------------------------------------------------------------
__global__ void __launch_bounds__(256)
k_scores(const float* __restrict__ Wh, const float* __restrict__ a,
         float* __restrict__ s1, float* __restrict__ s2)
{
  const int wave = threadIdx.x >> 5, lane = threadIdx.x & 31;
  const int row  = blockIdx.x * 8 + wave;
  float d1 = 0.f, d2 = 0.f;
  for (int k = lane; k < OUTF; k += 32) {
    float w = Wh[row * OUTF + k];
    d1 += w * a[k];
    d2 += w * a[OUTF + k];
  }
  #pragma unroll
  for (int off = 16; off; off >>= 1) {
    d1 += __shfl_xor(d1, off, 32);
    d2 += __shfl_xor(d2, off, 32);
  }
  if (lane == 0) { s1[row] = d1; s2[row] = d2; }
}

// ---------------------------------------------------------------------------
// Kernel 3: global max of s2 (upper-bound softmax shift; leaky_relu monotone).
// ---------------------------------------------------------------------------
__global__ void __launch_bounds__(256)
k_s2max(const float* __restrict__ s2, float* __restrict__ s2max)
{
  __shared__ float sm[256];
  float m = -3.0e38f;
  for (int i = threadIdx.x; i < NN; i += 256) m = fmaxf(m, s2[i]);
  sm[threadIdx.x] = m; __syncthreads();
  for (int s = 128; s; s >>= 1) {
    if (threadIdx.x < s) sm[threadIdx.x] = fmaxf(sm[threadIdx.x], sm[threadIdx.x + s]);
    __syncthreads();
  }
  if (threadIdx.x == 0) *s2max = sm[0];
}

// ---------------------------------------------------------------------------
// Kernel 4: single-pass fused masked softmax + aggregation + ELU.
// Block = 16 rows; for each 32-wide j-chunk: build P=exp(e-m_i) tile (bf16)
// in LDS cooperatively, then every wave runs 2x V_WMMA_F32_16X16X32_BF16
// (P-tile shared across waves, B from bf16 Wh^T with contiguous K).
// m_i = leaky(s1_i + max_j s2_j) is a valid shift => no online rescaling.
// ---------------------------------------------------------------------------
__global__ void __launch_bounds__(256)
k_gat(const int* __restrict__ adj, const unsigned short* __restrict__ WhbT,
      const float* __restrict__ s1, const float* __restrict__ s2,
      const float* __restrict__ s2maxp, float* __restrict__ out)
{
  __shared__ unsigned int Plu[16 * 17];   // 16 rows x 16 bf16-pairs, stride 17 => conflict-free
  __shared__ float Zp[256];
  __shared__ float Zs[16];

  const int tid  = threadIdx.x;
  const int i0   = blockIdx.x * 16;
  const int wave = tid >> 5, lane = tid & 31;
  const int lm   = lane & 15, hi = lane >> 4;
  const int prow = tid >> 4;              // P-gen: row 0..15
  const int pcol = (tid & 15) * 2;        // P-gen: col pair

  const float s1v = s1[i0 + prow];
  float mi = s1v + *s2maxp;
  mi = mi > 0.f ? mi : LRA * mi;
  float zpart = 0.f;

  const int n0 = wave * 32;
  v8f acc0 = {}; v8f acc1 = {};

  const long arow   = (long)(i0 + prow) * NN;
  const long bbase0 = (long)(n0 + lm) * NN + hi * 16;       // B frag: K contiguous
  const long bbase1 = (long)(n0 + 16 + lm) * NN + hi * 16;
  const int  abase  = lm * 17 + hi * 8;                     // A frag LDS base (u32)

  for (int jb = 0; jb < NN; jb += 32) {
    // ---- P tile: masked exp(e - m_i), coalesced adj stream ----
    int2 av = *(const int2*)(adj + arow + jb + pcol);
    __builtin_prefetch(adj + arow + jb + pcol + 4096, 0, 1); // global_prefetch_b8
    float e0 = s1v + s2[jb + pcol];
    float e1 = s1v + s2[jb + pcol + 1];
    e0 = e0 > 0.f ? e0 : LRA * e0;
    e1 = e1 > 0.f ? e1 : LRA * e1;
    float p0 = (av.x > 0) ? __expf(e0 - mi) : 0.f;  // exp(NEG_BIG) == 0 exactly
    float p1 = (av.y > 0) ? __expf(e1 - mi) : 0.f;
    zpart += p0 + p1;
    Plu[prow * 17 + (pcol >> 1)] = (unsigned int)f2bf(p0) | ((unsigned int)f2bf(p1) << 16);
    __syncthreads();

    // ---- A frag (shared P tile): lanes0-15 K 0-15; lanes16-31 K 16-31
    union { unsigned int u[8]; v16bf v; } af;
    #pragma unroll
    for (int t = 0; t < 4; ++t) {
      af.u[t]     = Plu[abase + t];
      af.u[t + 4] = Plu[abase + 4 + t];
    }
    // ---- B frags: 2x b128 per lane from Wh^T (bf16) ----
    union { uint4 q[2]; v16bf v; } b0, b1;
    b0.q[0] = *(const uint4*)(WhbT + bbase0 + jb);
    b0.q[1] = *(const uint4*)(WhbT + bbase0 + jb + 8);
    b1.q[0] = *(const uint4*)(WhbT + bbase1 + jb);
    b1.q[1] = *(const uint4*)(WhbT + bbase1 + jb + 8);

    acc0 = __builtin_amdgcn_wmma_f32_16x16x32_bf16(false, af.v, false, b0.v, (short)0, acc0, false, false);
    acc1 = __builtin_amdgcn_wmma_f32_16x16x32_bf16(false, af.v, false, b1.v, (short)0, acc1, false, false);
    __syncthreads();
  }

  // ---- per-row softmax denominator ----
  Zp[tid] = zpart; __syncthreads();
  if (tid < 16) {
    float s = 0.f;
    #pragma unroll
    for (int j = 0; j < 16; ++j) s += Zp[tid * 16 + j];
    Zs[tid] = s;
  }
  __syncthreads();

  // ---- normalize, ELU, store ----
  #pragma unroll
  for (int r = 0; r < 8; ++r) {
    const int m = r + hi * 8;
    const float inv = 1.f / Zs[m];
    float vA = acc0[r] * inv;
    float vB = acc1[r] * inv;
    vA = vA > 0.f ? vA : __expf(vA) - 1.f;
    vB = vB > 0.f ? vB : __expf(vB) - 1.f;
    out[(long)(i0 + m) * OUTF + n0 + lm]      = vA;
    out[(long)(i0 + m) * OUTF + n0 + 16 + lm] = vB;
  }
}

// ---------------------------------------------------------------------------
extern "C" void kernel_launch(void* const* d_in, const int* in_sizes, int n_in,
                              void* d_out, int out_size, void* d_ws, size_t ws_size,
                              hipStream_t stream) {
  const float* h   = (const float*)d_in[0];
  const int*   adj = (const int*)d_in[1];
  const float* W   = (const float*)d_in[2];
  const float* a   = (const float*)d_in[3];
  float* out = (float*)d_out;

  // Workspace layout (~12.1 MiB total):
  char* ws = (char*)d_ws;
  float*          Wh   = (float*)(ws);                        // 8 MiB  fp32 Wh
  unsigned short* WhbT = (unsigned short*)(ws + (8u << 20));  // 4 MiB  bf16 Wh^T
  float*          s1   = (float*)(ws + (12u << 20));          // 32 KiB
  float*          s2   = (float*)(ws + (12u << 20) + 32768);  // 32 KiB
  float*          s2m  = (float*)(ws + (12u << 20) + 65536);  // 4 B

  k_gemm_wh<<<NN / 16, 256, 0, stream>>>(h, W, Wh, WhbT);
  k_scores <<<NN / 8,  256, 0, stream>>>(Wh, a, s1, s2);
  k_s2max  <<<1,       256, 0, stream>>>(s2, s2m);
  k_gat    <<<NN / 16, 256, 0, stream>>>(adj, WhbT, s1, s2, s2m, out);
}
